// AnomalyAttention_55705725829124
// MI455X (gfx1250) — matrix operand order repaired and checked by reading, not compiled, not run on hardware
//
#include <hip/hip_runtime.h>
#include <hip/hip_bf16.h>
#include <math.h>

// ---------------- types ----------------
typedef __attribute__((ext_vector_type(16))) _Float16 v16h;
typedef __attribute__((ext_vector_type(8)))  _Float16 v8h;
typedef __attribute__((ext_vector_type(8)))  float    v8f;

#define BB 256
#define LL 100
#define HH 8
#define EE 64
#define DD 64
#define LP 112      // L padded to 7x16 tiles
#define SP 128      // S (K-dim of GEMM2) padded to 4x32
#define QK_STR 72   // halves per row (144B, 16B-aligned rows)
#define VT_STR 136  // halves per row (272B, 16B-aligned rows)
#define SH_STR 136

// A-operand (16x32 f16): lane m=lane&15 holds row m; half selects K sub-chunks
// elements 0..7  = K[kb + 8*half .. +7], elements 8..15 = K[kb+16+8*half .. +7]
static __device__ __forceinline__ v16h load_a(const _Float16* row, int kb, int half) {
  v8h lo = *(const v8h*)(row + kb + 8*half);
  v8h hi = *(const v8h*)(row + kb + 16 + 8*half);
  return __builtin_shufflevector(lo, hi, 0,1,2,3,4,5,6,7,8,9,10,11,12,13,14,15);
}
// B-operand (32x16 f16): lane n=lane&15 holds column n = contiguous source row;
// elements 0..7 = K[kb + 8*half .. +7], elements 8..15 = K[kb+16+8*half .. +7]
static __device__ __forceinline__ v16h load_b(const _Float16* row, int kb, int half) {
  v8h lo = *(const v8h*)(row + kb + 8*half);
  v8h hi = *(const v8h*)(row + kb + 16 + 8*half);
  return __builtin_shufflevector(lo, hi, 0,1,2,3,4,5,6,7,8,9,10,11,12,13,14,15);
}

__global__ __launch_bounds__(256)
void anomaly_attn_kernel(const float* __restrict__ q, const float* __restrict__ k,
                         const float* __restrict__ v,
                         float* __restrict__ oV, float* __restrict__ oSer,
                         float* __restrict__ oM)
{
  // region0 (Qh + Kh) is reused as Sh after GEMM1; Vt lives after it.
  __shared__ __align__(32) _Float16 lds[LP*QK_STR*2 + DD*VT_STR]; // 49,664 B
  _Float16* Qh = lds;
  _Float16* Kh = lds + LP*QK_STR;
  _Float16* Sh = lds;                  // alias: 112*136 <= 2*112*72
  _Float16* Vt = lds + LP*QK_STR*2;

  const int tid  = threadIdx.x;
  const int wave = tid >> 5;
  const int lane = tid & 31;
  const int n    = lane & 15;
  const int half = lane >> 4;
  const int bh   = blockIdx.x;
  const int b    = bh / HH, h = bh % HH;

  // ---- cooperative load: Q,K -> f16 (rows >= 100 zeroed) ----
  for (int idx = tid; idx < LP*EE; idx += 256) {
    int r = idx >> 6, e = idx & 63;
    _Float16 qv = (_Float16)0.0f, kv = (_Float16)0.0f;
    if (r < LL) {
      int g = ((b*LL + r)*HH + h)*EE + e;
      qv = (_Float16)q[g];
      kv = (_Float16)k[g];
    }
    Qh[r*QK_STR + e] = qv;
    Kh[r*QK_STR + e] = kv;
  }
  // ---- V -> f16 transposed Vt[d][s] (cols s >= 100 zeroed, up to 128) ----
  for (int idx = tid; idx < SP*DD; idx += 256) {
    int s = idx >> 6, d = idx & 63;             // coalesced global read of V rows
    _Float16 vv = (_Float16)0.0f;
    if (s < LL) vv = (_Float16)v[((b*LL + s)*HH + h)*DD + d];
    Vt[d*VT_STR + s] = vv;
  }
  __syncthreads();

  const v8f vzero = {0.f,0.f,0.f,0.f,0.f,0.f,0.f,0.f};
  v8f acc[7];
  #pragma unroll
  for (int t = 0; t < 7; t++) acc[t] = vzero;

  // ---- GEMM1: S = Q * K^T  (waves 0..6 each own a 16-row strip) ----
  if (wave < 7) {
    const int mbase = wave * 16;
    const _Float16* qrow = Qh + (mbase + n)*QK_STR;
    v16h a0 = load_a(qrow, 0,  half);
    v16h a1 = load_a(qrow, 32, half);
    #pragma unroll
    for (int t = 0; t < 7; t++) {
      const _Float16* krow = Kh + (t*16 + n)*QK_STR;
      v16h b0 = load_b(krow, 0,  half);
      v16h b1 = load_b(krow, 32, half);
      acc[t] = __builtin_amdgcn_wmma_f32_16x16x32_f16(false, a0, false, b0,
                                                      (short)0, acc[t], false, false);
      acc[t] = __builtin_amdgcn_wmma_f32_16x16x32_f16(false, a1, false, b1,
                                                      (short)0, acc[t], false, false);
    }
  }
  __syncthreads();               // Qh/Kh dead from here on

  // ---- zero Sh (covers s-padding 100..127 and row padding) ----
  for (int idx = tid; idx < LP*SH_STR; idx += 256) Sh[idx] = (_Float16)0.0f;
  __syncthreads();

  // ---- row stats (M1, M2), softmax, series writes ----
  if (wave < 7) {
    const int mbase = wave * 16;
    const size_t serbase = (size_t)bh * LL * LL;
    #pragma unroll
    for (int r = 0; r < 8; r++) {
      const int row = mbase + 8*half + r;
      float m1max = -1e30f, m1sum = 0.f, m2max = -1e30f, m2sum = 0.f;
      #pragma unroll
      for (int t = 0; t < 7; t++) {
        int col = t*16 + n;
        if (col < LL) {
          float val = acc[t][r];
          m1max = fmaxf(m1max, val);
          m1sum += val;
          int dd = row - col;
          float sp = (dd < 3 && dd > -3) ? 0.f : val;  // band |i-j|<3 -> 0
          m2max = fmaxf(m2max, sp);
          m2sum += sp;
        }
      }
      #pragma unroll
      for (int msk = 1; msk < 16; msk <<= 1) {   // reduce within 16-lane half
        m1max = fmaxf(m1max, __shfl_xor(m1max, msk, 32));
        m1sum += __shfl_xor(m1sum, msk, 32);
        m2max = fmaxf(m2max, __shfl_xor(m2max, msk, 32));
        m2sum += __shfl_xor(m2sum, msk, 32);
      }
      // softmax over attn = scale*scores (scale = 1/8)
      float pe[7];
      float psum = 0.f;
      #pragma unroll
      for (int t = 0; t < 7; t++) {
        int col = t*16 + n;
        float e = 0.f;
        if (col < LL) e = __expf(0.125f * (acc[t][r] - m1max));
        pe[t] = e;
        psum += e;
      }
      #pragma unroll
      for (int msk = 1; msk < 16; msk <<= 1) psum += __shfl_xor(psum, msk, 32);
      float inv = 1.0f / psum;

      if (row < LL) {
        #pragma unroll
        for (int t = 0; t < 7; t++) {
          int col = t*16 + n;
          if (col < LL) {
            float p = pe[t] * inv;
            if (col == row) p = 0.f;              // zero diagonal after softmax
            oSer[serbase + (size_t)row*LL + col] = p;
            Sh[row*SH_STR + col] = (_Float16)p;
          }
        }
        if (n == 0) {
          float M = (m1max - m1sum * (1.0f/LL)) - (m2max - m2sum * (1.0f/(LL - 6)));
          oM[(size_t)bh*LL + row] = M;
        }
      }
    }
  }
  __syncthreads();

  // ---- GEMM2: Out = series * V  (K-dim = s, 4 chunks of 32; padding is zero) ----
  if (wave < 7) {
    const int mbase = wave * 16;
    const _Float16* srow = Sh + (mbase + n)*SH_STR;
    v16h sa[4];
    #pragma unroll
    for (int kb = 0; kb < 4; kb++) sa[kb] = load_a(srow, kb*32, half);
    #pragma unroll
    for (int t = 0; t < 4; t++) {
      const _Float16* vrow = Vt + (t*16 + n)*VT_STR;
      v8f c = vzero;
      #pragma unroll
      for (int kb = 0; kb < 4; kb++) {
        v16h bm = load_b(vrow, kb*32, half);
        c = __builtin_amdgcn_wmma_f32_16x16x32_f16(false, sa[kb], false, bm,
                                                   (short)0, c, false, false);
      }
      #pragma unroll
      for (int r = 0; r < 8; r++) {
        int row = mbase + 8*half + r;
        if (row < LL)
          oV[(((size_t)b*LL + row)*HH + h)*DD + t*16 + n] = c[r];
      }
    }
  }
}

// ---- prior / sig: pure elementwise, bandwidth-bound ----
__global__ __launch_bounds__(256)
void prior_kernel(const float* __restrict__ sigma, const float* __restrict__ distances,
                  float* __restrict__ oPrior, float* __restrict__ oSig)
{
  const int bh = blockIdx.x;
  const int b = bh / HH, h = bh % HH;
  __shared__ float sigs[LL];
  for (int l = threadIdx.x; l < LL; l += 256) {
    float s0 = sigma[(b*LL + l)*HH + h];
    float sg = 1.0f / (1.0f + __expf(-5.0f * s0)) + 1e-5f;   // sigmoid(5x)+1e-5
    sg = __powf(3.0f, sg) - 1.0f;                            // 3^sg - 1
    sigs[l] = sg;
  }
  __syncthreads();
  const size_t base = (size_t)bh * LL * LL;
  for (int idx = threadIdx.x; idx < LL*LL; idx += 256) {
    int l = idx / LL;
    float sg = sigs[l];
    float d  = distances[idx];
    float p  = 0.3989422804014327f / sg * __expf(-0.5f * d * d / (sg * sg));
    oPrior[base + idx] = p;
    oSig[base + idx]   = sg;
  }
}

extern "C" void kernel_launch(void* const* d_in, const int* in_sizes, int n_in,
                              void* d_out, int out_size, void* d_ws, size_t ws_size,
                              hipStream_t stream) {
  const float* q     = (const float*)d_in[0];
  const float* k     = (const float*)d_in[1];
  const float* v     = (const float*)d_in[2];
  const float* sigma = (const float*)d_in[3];
  const float* dist  = (const float*)d_in[4];
  // d_in[5] = attn_mask, unused (mask_flag=False)

  float* out    = (float*)d_out;
  float* oV     = out;                         // (B,L,H,D)   13,107,200
  float* oSer   = out + 13107200;              // (B,H,L,L)   20,480,000
  float* oPrior = out + 33587200;              // (B,H,L,L)   20,480,000
  float* oSig   = out + 54067200;              // (B,H,L,L)   20,480,000
  float* oM     = out + 74547200;              // (B,H,L)        204,800

  anomaly_attn_kernel<<<BB*HH, 256, 0, stream>>>(q, k, v, oV, oSer, oM);
  prior_kernel<<<BB*HH, 256, 0, stream>>>(sigma, dist, oPrior, oSig);
}
